// ContrastiveLoss_90692529422675
// MI455X (gfx1250) — matrix operand order repaired and checked by reading, not compiled, not run on hardware
//
#include <hip/hip_runtime.h>
#include <hip/hip_bf16.h>

typedef __attribute__((ext_vector_type(16))) _Float16     v16h;
typedef __attribute__((ext_vector_type(4)))  _Float16     v4h;
typedef __attribute__((ext_vector_type(8)))  float        v8f;
typedef __attribute__((ext_vector_type(8)))  unsigned int v8u;

#define N_TOT 8192      // VIEWS * B
#define BHALF 4096      // B
#define DIM   128       // D
#define INV_TEMP 10.0f  // 1 / temperature
#define NEG_BIG (-1.0e30f)

// ---------------------------------------------------------------------------
// Kernel 0: zero the scalar accumulator (harness poisons d_out).
// ---------------------------------------------------------------------------
__global__ void zero_out_kernel(float* out) {
    if (threadIdx.x == 0 && blockIdx.x == 0) out[0] = 0.0f;
}

// ---------------------------------------------------------------------------
// Kernel 1: row-normalize [N_TOT, DIM] f32 -> f16 in workspace.
// One wave (32 lanes) per row; each lane owns 4 consecutive features.
// ---------------------------------------------------------------------------
__global__ __launch_bounds__(256) void normalize_f16_kernel(
    const float* __restrict__ emb0, const float* __restrict__ emb1,
    _Float16* __restrict__ zn) {
    const int lane = threadIdx.x & 31;
    const int row  = (int)((blockIdx.x * blockDim.x + threadIdx.x) >> 5);
    if (row >= N_TOT) return;

    const float* src = (row < BHALF) ? (emb0 + (size_t)row * DIM)
                                     : (emb1 + (size_t)(row - BHALF) * DIM);
    float4 x = ((const float4*)src)[lane];                 // features lane*4 .. lane*4+3
    float ss = x.x * x.x + x.y * x.y + x.z * x.z + x.w * x.w;
#pragma unroll
    for (int mask = 16; mask >= 1; mask >>= 1)
        ss += __shfl_xor(ss, mask, 32);
    const float inv = 1.0f / fmaxf(sqrtf(ss), 1e-8f);

    v4h h;
    h[0] = (_Float16)(x.x * inv);
    h[1] = (_Float16)(x.y * inv);
    h[2] = (_Float16)(x.z * inv);
    h[3] = (_Float16)(x.w * inv);
    ((v4h*)(zn + (size_t)row * DIM))[lane] = h;            // 8-byte store
}

// ---------------------------------------------------------------------------
// Kernel 2: fused Zn*Zn^T (WMMA f16) + masked online log-sum-exp + positive
// extraction. One wave handles 16 rows and sweeps all 8192 columns using a
// 4-deep rotating B-fragment buffer (prefetch distance 2 tiles), so every
// tile's 8x b128 loads are issued two slots before their WMMAs consume them.
//
// VGPR layouts per CDNA5 ISA 7.12.2 (wave32):
//  A (16x32 f16, M x K):  lane = half*16 + m;  VGPR v holds K = kc*32 +
//                         (v>>2)*16 + (v&3)*2 + half*8  (pair, 4B aligned)
//  B (32x16 f16, K x N):  lane = khalf*16 + n; VGPR v holds K = kc*32 +
//                         (v>>2)*16 + (v&3)*2 + khalf*8 for column n
//  C/D (16x16 f32):       lane = half*16 + n;  element v -> row m = v + 8*half
// ---------------------------------------------------------------------------
__global__ __launch_bounds__(256) void simclr_loss_kernel(
    const _Float16* __restrict__ zn, float* __restrict__ out) {
    const int lane = threadIdx.x & 31;
    const int wave = (int)((blockIdx.x * blockDim.x + threadIdx.x) >> 5);
    const int i0   = wave * 16;          // first row of this wave's tile
    const int half = lane >> 4;
    const int nl   = lane & 15;

    // ---- preload A fragments for all 4 K-chunks (row block i0..i0+15) ----
    v16h A[4];
    {
        const _Float16* arow = zn + (size_t)(i0 + nl) * DIM;  // A lane row: m = nl
#pragma unroll
        for (int kc = 0; kc < 4; ++kc) {
            v8u au;
#pragma unroll
            for (int v = 0; v < 8; ++v) {
                const int klo = kc * 32 + ((v >> 2) << 4) + ((v & 3) << 1) + (half << 3);
                au[v] = *(const unsigned int*)(arow + klo);
            }
            A[kc] = __builtin_bit_cast(v16h, au);
        }
    }

    // ---- per-lane online softmax state: 8 (row, column-class) streams ----
    float mx[8], sm[8], pv[8];
    int   rrow[8], pcol[8];
#pragma unroll
    for (int v = 0; v < 8; ++v) {
        mx[v]   = -__builtin_inff();
        sm[v]   = 0.0f;
        pv[v]   = 0.0f;
        rrow[v] = i0 + v + (half << 3);                       // C-layout row
        pcol[v] = (rrow[v] < BHALF) ? rrow[v] + BHALF : rrow[v] - BHALF;
    }

    // ---- B-fragment fetch (8x b128 from L2) ----
    auto load_tile = [&](int j0, v8u (&bu)[4]) {
        const _Float16* brow = zn + (size_t)(j0 + nl) * DIM;  // B lane col: n = nl
#pragma unroll
        for (int kc = 0; kc < 4; ++kc)
#pragma unroll
            for (int v = 0; v < 8; ++v)
                bu[kc][v] = *(const unsigned int*)(brow + kc * 32 + ((v >> 2) << 4) + ((v & 3) << 1) + (half << 3));
    };

    // ---- 4x WMMA + masked online-LSE epilogue for one 16x16 tile ----
    auto process_tile = [&](int j0, v8u (&bu)[4]) {
        v8f acc = {};
#pragma unroll
        for (int kc = 0; kc < 4; ++kc)
            acc = __builtin_amdgcn_wmma_f32_16x16x32_f16(
                false, A[kc], false, __builtin_bit_cast(v16h, bu[kc]),
                (short)0, acc, false, false);

        const int c = j0 + nl;                                // this lane's column
#pragma unroll
        for (int v = 0; v < 8; ++v) {
            float val = acc[v] * INV_TEMP;
            val = (c == rrow[v]) ? NEG_BIG : val;             // drop diagonal
            const float nm = fmaxf(mx[v], val);               // branchless online LSE
            sm[v] = sm[v] * __expf(mx[v] - nm) + __expf(val - nm);
            mx[v] = nm;
            pv[v] += (c == pcol[v]) ? val : 0.0f;             // positive logit
        }
    };

    // ---- software-pipelined sweep: 512 tiles, 4 per iteration, rotating
    //      buffers with prefetch distance 2 (wrap-around loads are harmless) ----
    v8u b0[4], b1[4], b2[4], b3[4];
    load_tile(0,  b0);
    load_tile(16, b1);
    for (int j0 = 0; j0 < N_TOT; j0 += 64) {
        load_tile(j0 + 32, b2);
        process_tile(j0, b0);
        load_tile(j0 + 48, b3);
        process_tile(j0 + 16, b1);
        load_tile((j0 + 64) & (N_TOT - 1), b0);
        process_tile(j0 + 32, b2);
        load_tile((j0 + 80) & (N_TOT - 1), b1);
        process_tile(j0 + 48, b3);
    }

    // ---- combine the 16 lane-streams of each half (xor masks stay in-half) ----
    float contrib = 0.0f;
#pragma unroll
    for (int v = 0; v < 8; ++v) {
        float m = mx[v], s = sm[v], p = pv[v];
#pragma unroll
        for (int mask = 8; mask >= 1; mask >>= 1) {
            const float mo = __shfl_xor(m, mask, 32);
            const float so = __shfl_xor(s, mask, 32);
            const float po = __shfl_xor(p, mask, 32);
            const float nm = fmaxf(m, mo);
            s = s * __expf(m - nm) + so * __expf(mo - nm);
            m = nm;
            p += po;
        }
        contrib += (m + __logf(s)) - p;                       // lse_row - pos_row
    }
    if (nl == 0)                                              // lanes 0 and 16
        atomicAdd(out, contrib * (1.0f / (float)N_TOT));
}

// ---------------------------------------------------------------------------
extern "C" void kernel_launch(void* const* d_in, const int* in_sizes, int n_in,
                              void* d_out, int out_size, void* d_ws, size_t ws_size,
                              hipStream_t stream) {
    (void)in_sizes; (void)n_in; (void)out_size; (void)ws_size;
    const float* emb0 = (const float*)d_in[0];
    const float* emb1 = (const float*)d_in[1];
    float*       out  = (float*)d_out;
    _Float16*    zn   = (_Float16*)d_ws;   // N_TOT * DIM * 2 bytes = 2 MB

    zero_out_kernel<<<1, 64, 0, stream>>>(out);
    // 8192 rows, 1 wave each, 8 waves/block -> 1024 blocks
    normalize_f16_kernel<<<(N_TOT * 32) / 256, 256, 0, stream>>>(emb0, emb1, zn);
    // 512 row-tiles, 1 wave each, 8 waves/block -> 64 blocks
    simclr_loss_kernel<<<((N_TOT / 16) * 32) / 256, 256, 0, stream>>>(zn, out);
}
